// QEMini_88407606820890
// MI455X (gfx1250) — matrix operand order repaired and checked by reading, not compiled, not run on hardware
//
#include <hip/hip_runtime.h>

// Model dims (fixed by reference)
#define D_    1024
#define H_    16
#define DK_   64
#define L_    12
#define FF_   4096
#define B_    2
#define S_    512
#define PAD_  1

// GEMM tiling
#define BM 128
#define BKK 32
#define LDSK 40          // padded K stride (ushorts) for LDS tiles
#define GT 256           // 8 wave32 waves

typedef __attribute__((ext_vector_type(16))) __bf16 v16bf;
typedef __attribute__((ext_vector_type(8)))  float  v8f;
typedef __attribute__((ext_vector_type(8)))  unsigned short v8us;
typedef __attribute__((ext_vector_type(16))) unsigned short v16us;
typedef __attribute__((ext_vector_type(4)))  unsigned int   v4u;

union BfVec { v16us us; v16bf bf; };

enum { EPI_NONE = 0, EPI_SCALE, EPI_GELU, EPI_RELU, EPI_RESID, EPI_MASK };

__device__ __forceinline__ unsigned short f2bf(float f) {
  unsigned int u = __float_as_uint(f);
  u += 0x7FFFu + ((u >> 16) & 1u);   // round-to-nearest-even
  return (unsigned short)(u >> 16);
}

// Pack two fp32 into one dword of bf16 pair.
// Round-to-nearest (ties away from zero): +0x8000 then keep high 16 bits;
// both high halves packed with a single V_PERM_B32.
__device__ __forceinline__ unsigned int pack2bf(float lo, float hi) {
  const unsigned int ulo = __float_as_uint(lo) + 0x8000u;
  const unsigned int uhi = __float_as_uint(hi) + 0x8000u;
#if __has_builtin(__builtin_amdgcn_perm)
  // concatenated {uhi,ulo}: bytes 0-3 = ulo, 4-7 = uhi; pick [lo2,lo3,hi2,hi3]
  return __builtin_amdgcn_perm(uhi, ulo, 0x07060302u);
#else
  return (uhi & 0xFFFF0000u) | (ulo >> 16);
#endif
}

// ---------------------------------------------------------------------------
// Fully tile-divisible batched GEMM (no bounds checks anywhere):
//   C[z] = epi( A[z] (MxK) * B[z] (KxN) + bias )
// Requirements: M % 128 == 0, N % TBN == 0, K % 32 == 0.
// Software-pipelined: next K-tile's global loads are issued before the WMMA
// block so memory latency overlaps matrix math.
// ---------------------------------------------------------------------------
template <int EPI, int TBN, bool TRANSB>
__global__ __launch_bounds__(GT) void gemm_full(
    const float* __restrict__ A, long long sAo, long long sAi, int lda,
    const float* __restrict__ Bm, long long sBo, long long sBi, int ldb,
    float* __restrict__ C, long long sCo, long long sCi, int ldc,
    const float* __restrict__ bias,
    const float* __restrict__ resid, long long sRo, long long sRi, int ldr,
    const int* __restrict__ mask, int maskStride,
    float alpha, int K, int innerB)
{
  constexpr int NT   = TBN / 32;   // WMMA n-tiles per wave
  constexpr int WNSZ = TBN / 2;    // columns per wave-column group
  constexpr int NCW  = TBN / 16;   // floats per thread-row (non-trans B staging)
  constexpr int NBV  = NCW / 4;    // float4 loads per row

  __shared__ unsigned short As[BM * LDSK];
  __shared__ unsigned short Bs[TBN * LDSK];

  const int z  = blockIdx.z;
  const int bi = z / innerB;
  const int hi = z - bi * innerB;
  A  += bi * sAo + hi * sAi;
  Bm += bi * sBo + hi * sBi;
  C  += bi * sCo + hi * sCi;
  if (EPI == EPI_RESID) resid += bi * sRo + hi * sRi;

  const int m0   = blockIdx.y * BM;
  const int n0   = blockIdx.x * TBN;
  const int tid  = threadIdx.x;
  const int lane = tid & 31;
  const int wave = tid >> 5;
  const int wm   = wave & 3;    // 4 row groups of 32
  const int wn   = wave >> 2;   // 2 col groups of WNSZ

  // staging-thread mappings
  const int sm  = tid >> 1;          // A / trans-B: tile row (0..127)
  const int skq = (tid & 1) * 16;    // A / trans-B: k sub-chunk (0 or 16)
  const int kp  = tid >> 4;          // non-trans B: k pair (0..15) -> k = 2*kp
  const int nc  = (tid & 15) * NCW;  // non-trans B: n chunk base

  const float* aPtr  = A + (long long)(m0 + sm) * lda + skq;
  const float* bPtrT = Bm + (long long)(n0 + sm) * ldb + skq;        // TRANSB
  const float* bPtrN = Bm + (long long)(2 * kp) * ldb + n0 + nc;     // !TRANSB

  float2 ra[8];
  float4 rbT[4];
  float4 rb0[NBV], rb1[NBV];

  // initial global loads (tile k0 = 0)
  #pragma unroll
  for (int j = 0; j < 8; ++j) ra[j] = ((const float2*)aPtr)[j];
  if constexpr (TRANSB) {
    #pragma unroll
    for (int j = 0; j < 4; ++j) rbT[j] = ((const float4*)bPtrT)[j];
  } else {
    #pragma unroll
    for (int j = 0; j < NBV; ++j) {
      rb0[j] = ((const float4*)bPtrN)[j];
      rb1[j] = ((const float4*)(bPtrN + ldb))[j];
    }
  }

  v8f acc[2][NT];
  const v8f vzero = {0.f, 0.f, 0.f, 0.f, 0.f, 0.f, 0.f, 0.f};
  #pragma unroll
  for (int i = 0; i < 2; ++i)
    #pragma unroll
    for (int j = 0; j < NT; ++j) acc[i][j] = vzero;

  for (int k0 = 0; k0 < K; k0 += BKK) {
    // ---- registers -> LDS (packed bf16, 16B stores) ----
    {
      v4u p0, p1;
      #pragma unroll
      for (int j = 0; j < 4; ++j) {
        p0[j] = pack2bf(ra[j].x, ra[j].y);
        p1[j] = pack2bf(ra[j + 4].x, ra[j + 4].y);
      }
      *(v4u*)&As[sm * LDSK + skq]     = p0;
      *(v4u*)&As[sm * LDSK + skq + 8] = p1;
    }
    if constexpr (TRANSB) {
      const float* f = (const float*)rbT;
      v4u p0, p1;
      #pragma unroll
      for (int j = 0; j < 4; ++j) {
        p0[j] = pack2bf(f[2 * j], f[2 * j + 1]);
        p1[j] = pack2bf(f[2 * j + 8], f[2 * j + 9]);
      }
      *(v4u*)&Bs[sm * LDSK + skq]     = p0;
      *(v4u*)&Bs[sm * LDSK + skq + 8] = p1;
    } else {
      const float* f0 = (const float*)rb0;
      const float* f1 = (const float*)rb1;
      #pragma unroll
      for (int j = 0; j < NCW; ++j)
        *(unsigned int*)&Bs[(nc + j) * LDSK + 2 * kp] = pack2bf(f0[j], f1[j]);
    }
    __syncthreads();

    // ---- issue next tile's global loads (overlap with WMMA below) ----
    if (k0 + BKK < K) {
      aPtr += BKK;
      #pragma unroll
      for (int j = 0; j < 8; ++j) ra[j] = ((const float2*)aPtr)[j];
      if constexpr (TRANSB) {
        bPtrT += BKK;
        #pragma unroll
        for (int j = 0; j < 4; ++j) rbT[j] = ((const float4*)bPtrT)[j];
      } else {
        bPtrN += (long long)BKK * ldb;
        #pragma unroll
        for (int j = 0; j < NBV; ++j) {
          rb0[j] = ((const float4*)bPtrN)[j];
          rb1[j] = ((const float4*)(bPtrN + ldb))[j];
        }
      }
    }

    // ---- fragments + WMMA ----
    // A 16x32 bf16: lanes 0-15 (rows 0..15): K = {0..7, 16..23}; lanes 16-31: {8..15, 24..31}
    const int rsel  = lane & 15;
    const int baseA = (lane >> 4) * 8;
    // B 32x16 bf16: lane = column; lanes 0-15: K = {0..7, 16..23}; lanes 16-31: {8..15, 24..31}
    const int baseB = (lane >> 4) * 8;

    BfVec afr[2], bfr[NT];
    #pragma unroll
    for (int mt = 0; mt < 2; ++mt) {
      const int row = wm * 32 + mt * 16 + rsel;
      v8us lo = *(const v8us*)&As[row * LDSK + baseA];
      v8us hv = *(const v8us*)&As[row * LDSK + 16 + baseA];
      afr[mt].us = __builtin_shufflevector(lo, hv, 0,1,2,3,4,5,6,7,8,9,10,11,12,13,14,15);
    }
    #pragma unroll
    for (int nt = 0; nt < NT; ++nt) {
      const int row = wn * WNSZ + nt * 16 + rsel;
      v8us lo = *(const v8us*)&Bs[row * LDSK + baseB];
      v8us hv = *(const v8us*)&Bs[row * LDSK + baseB + 16];
      bfr[nt].us = __builtin_shufflevector(lo, hv, 0,1,2,3,4,5,6,7,8,9,10,11,12,13,14,15);
    }
    #pragma unroll
    for (int mt = 0; mt < 2; ++mt)
      #pragma unroll
      for (int nt = 0; nt < NT; ++nt)
        acc[mt][nt] = __builtin_amdgcn_wmma_f32_16x16x32_bf16(
            false, afr[mt].bf, false, bfr[nt].bf, (short)0, acc[mt][nt], false, false);
    __syncthreads();
  }

  // ---- epilogue (no guards) ----
  #pragma unroll
  for (int mt = 0; mt < 2; ++mt) {
    const int mbase = m0 + wm * 32 + mt * 16 + ((lane >> 4) << 3);
    #pragma unroll
    for (int nt = 0; nt < NT; ++nt) {
      const int gn = n0 + wn * WNSZ + nt * 16 + (lane & 15);
      const float bv = bias ? bias[gn] : 0.0f;
      #pragma unroll
      for (int r = 0; r < 8; ++r) {
        const int gm = mbase + r;
        float x = acc[mt][nt][r] + bv;
        if (EPI == EPI_SCALE) {
          x *= alpha;
        } else if (EPI == EPI_GELU) {
          x = 0.5f * x * (1.0f + erff(x * 0.70710678118654752440f));
        } else if (EPI == EPI_RELU) {
          x = fmaxf(x, 0.0f);
        } else if (EPI == EPI_RESID) {
          x += resid[(long long)gm * ldr + gn];
        } else if (EPI == EPI_MASK) {
          const int* mr = mask + (long long)bi * maskStride;
          if (mr[gn] == 0 || mr[gm] == 0) x += -1000000000.0f;
        }
        C[(long long)gm * ldc + gn] = x;
      }
    }
  }
}

// ---------------------------------------------------------------------------
// Guarded generic GEMM (used only for the tiny classifier GEMM, M = 2).
// ---------------------------------------------------------------------------
template <int EPI>
__global__ __launch_bounds__(GT) void gemm_part(
    const float* __restrict__ A, int lda,
    const float* __restrict__ Bm, int ldb,
    float* __restrict__ C, int ldc,
    const float* __restrict__ bias,
    int M, int N, int K)
{
  __shared__ unsigned short As[BM * LDSK];
  __shared__ unsigned short Bs[128 * LDSK];

  const int m0   = blockIdx.y * BM;
  const int n0   = blockIdx.x * 128;
  const int tid  = threadIdx.x;
  const int lane = tid & 31;
  const int wave = tid >> 5;
  const int wm   = wave & 3;
  const int wn   = wave >> 2;

  const int sm  = tid >> 1;
  const int skq = (tid & 1) * 16;

  v8f acc[2][4];
  const v8f vzero = {0.f, 0.f, 0.f, 0.f, 0.f, 0.f, 0.f, 0.f};
  #pragma unroll
  for (int i = 0; i < 2; ++i)
    #pragma unroll
    for (int j = 0; j < 4; ++j) acc[i][j] = vzero;

  for (int k0 = 0; k0 < K; k0 += BKK) {
    {
      const int gm = m0 + sm;
      #pragma unroll
      for (int i = 0; i < 16; ++i) {
        const int gk = k0 + skq + i;
        float v = (gm < M && gk < K) ? A[(long long)gm * lda + gk] : 0.0f;
        As[sm * LDSK + skq + i] = f2bf(v);
      }
    }
    {
      const int n2  = tid & 127;
      const int kq2 = (tid >> 7) * 16;
      const int gn  = n0 + n2;
      #pragma unroll
      for (int i = 0; i < 16; ++i) {
        const int gk = k0 + kq2 + i;
        float v = (gn < N && gk < K) ? Bm[(long long)gk * ldb + gn] : 0.0f;
        Bs[n2 * LDSK + kq2 + i] = f2bf(v);
      }
    }
    __syncthreads();

    const int rsel  = lane & 15;
    const int baseA = (lane >> 4) * 8;
    const int baseB = (lane >> 4) * 8;

    BfVec afr[2], bfr[4];
    #pragma unroll
    for (int mt = 0; mt < 2; ++mt) {
      const int row = wm * 32 + mt * 16 + rsel;
      v8us lo = *(const v8us*)&As[row * LDSK + baseA];
      v8us hv = *(const v8us*)&As[row * LDSK + 16 + baseA];
      afr[mt].us = __builtin_shufflevector(lo, hv, 0,1,2,3,4,5,6,7,8,9,10,11,12,13,14,15);
    }
    #pragma unroll
    for (int nt = 0; nt < 4; ++nt) {
      const int row = wn * 64 + nt * 16 + rsel;
      v8us lo = *(const v8us*)&Bs[row * LDSK + baseB];
      v8us hv = *(const v8us*)&Bs[row * LDSK + baseB + 16];
      bfr[nt].us = __builtin_shufflevector(lo, hv, 0,1,2,3,4,5,6,7,8,9,10,11,12,13,14,15);
    }
    #pragma unroll
    for (int mt = 0; mt < 2; ++mt)
      #pragma unroll
      for (int nt = 0; nt < 4; ++nt)
        acc[mt][nt] = __builtin_amdgcn_wmma_f32_16x16x32_bf16(
            false, afr[mt].bf, false, bfr[nt].bf, (short)0, acc[mt][nt], false, false);
    __syncthreads();
  }

  #pragma unroll
  for (int mt = 0; mt < 2; ++mt) {
    const int mbase = m0 + wm * 32 + mt * 16 + ((lane >> 4) << 3);
    #pragma unroll
    for (int nt = 0; nt < 4; ++nt) {
      const int gn = n0 + wn * 64 + nt * 16 + (lane & 15);
      const float bv = (bias && gn < N) ? bias[gn] : 0.0f;
      #pragma unroll
      for (int r = 0; r < 8; ++r) {
        const int gm = mbase + r;
        if (gm < M && gn < N) {
          float x = acc[mt][nt][r] + bv;
          if (EPI == EPI_RELU) x = fmaxf(x, 0.0f);
          C[(long long)gm * ldc + gn] = x;
        }
      }
    }
  }
}

// ---------------------------------------------------------------------------
// pos_ids = cumsum(ids != PAD) * (ids != PAD) + PAD
// ---------------------------------------------------------------------------
__global__ void posids_kernel(const int* __restrict__ ids, int* __restrict__ pos)
{
  const int b = threadIdx.x;
  if (b < B_) {
    int run = 0;
    for (int s = 0; s < S_; ++s) {
      const int np = (ids[b * S_ + s] != PAD_) ? 1 : 0;
      run += np;
      pos[b * S_ + s] = run * np + PAD_;
    }
  }
}

// ---------------------------------------------------------------------------
// Fused embedding gather + layernorm; one block per token.
// ---------------------------------------------------------------------------
__global__ __launch_bounds__(256) void embed_ln_kernel(
    const int* __restrict__ ids, const int* __restrict__ pos,
    const float* __restrict__ we, const float* __restrict__ pe,
    const float* __restrict__ te, const float* __restrict__ g,
    const float* __restrict__ b, float* __restrict__ out, float eps)
{
  __shared__ float red[256];
  const int tok = blockIdx.x;
  const int tid = threadIdx.x;
  const long long wid = (long long)ids[tok] * D_;
  const long long pid = (long long)pos[tok] * D_;
  float v[4];
  float s = 0.f;
  #pragma unroll
  for (int i = 0; i < 4; ++i) {
    const int c = tid + 256 * i;
    v[i] = we[wid + c] + pe[pid + c] + te[c];
    s += v[i];
  }
  red[tid] = s; __syncthreads();
  for (int o = 128; o > 0; o >>= 1) { if (tid < o) red[tid] += red[tid + o]; __syncthreads(); }
  const float mean = red[0] * (1.0f / D_);
  __syncthreads();
  float sq = 0.f;
  #pragma unroll
  for (int i = 0; i < 4; ++i) { const float d = v[i] - mean; sq += d * d; }
  red[tid] = sq; __syncthreads();
  for (int o = 128; o > 0; o >>= 1) { if (tid < o) red[tid] += red[tid + o]; __syncthreads(); }
  const float inv = rsqrtf(red[0] * (1.0f / D_) + eps);
  #pragma unroll
  for (int i = 0; i < 4; ++i) {
    const int c = tid + 256 * i;
    out[(long long)tok * D_ + c] = (v[i] - mean) * inv * g[c] + b[c];
  }
}

// ---------------------------------------------------------------------------
// LayerNorm over D; optional mirror store (hidden_states output).
// ---------------------------------------------------------------------------
__global__ __launch_bounds__(256) void layernorm_kernel(
    const float* __restrict__ x, const float* __restrict__ g,
    const float* __restrict__ b, float* __restrict__ out,
    float* __restrict__ out2, float eps)
{
  __shared__ float red[256];
  const long long row = blockIdx.x;
  const int tid = threadIdx.x;
  const float* xr = x + row * D_;
  float v[4];
  float s = 0.f;
  #pragma unroll
  for (int i = 0; i < 4; ++i) { v[i] = xr[tid + 256 * i]; s += v[i]; }
  red[tid] = s; __syncthreads();
  for (int o = 128; o > 0; o >>= 1) { if (tid < o) red[tid] += red[tid + o]; __syncthreads(); }
  const float mean = red[0] * (1.0f / D_);
  __syncthreads();
  float sq = 0.f;
  #pragma unroll
  for (int i = 0; i < 4; ++i) { const float d = v[i] - mean; sq += d * d; }
  red[tid] = sq; __syncthreads();
  for (int o = 128; o > 0; o >>= 1) { if (tid < o) red[tid] += red[tid + o]; __syncthreads(); }
  const float inv = rsqrtf(red[0] * (1.0f / D_) + eps);
  #pragma unroll
  for (int i = 0; i < 4; ++i) {
    const int c = tid + 256 * i;
    const float o2 = (v[i] - mean) * inv * g[c] + b[c];
    out[row * D_ + c] = o2;
    if (out2) out2[row * D_ + c] = o2;
  }
}

// ---------------------------------------------------------------------------
// In-place softmax over S=512 scores + log_softmax of the resulting probs.
// ---------------------------------------------------------------------------
__global__ __launch_bounds__(256) void softmax2_kernel(
    float* __restrict__ attn, float* __restrict__ lat)
{
  __shared__ float red[256];
  const long long row = blockIdx.x;
  const int tid = threadIdx.x;
  float* pr = attn + row * S_;
  float* lr = lat + row * S_;
  const float v0 = pr[tid], v1 = pr[tid + 256];

  red[tid] = fmaxf(v0, v1); __syncthreads();
  for (int o = 128; o > 0; o >>= 1) { if (tid < o) red[tid] = fmaxf(red[tid], red[tid + o]); __syncthreads(); }
  const float mx = red[0]; __syncthreads();

  const float e0 = expf(v0 - mx), e1 = expf(v1 - mx);
  red[tid] = e0 + e1; __syncthreads();
  for (int o = 128; o > 0; o >>= 1) { if (tid < o) red[tid] += red[tid + o]; __syncthreads(); }
  const float inv = 1.0f / red[0]; __syncthreads();

  const float p0 = e0 * inv, p1 = e1 * inv;
  pr[tid] = p0; pr[tid + 256] = p1;

  // log_softmax applied to the probabilities (as the reference does)
  red[tid] = fmaxf(p0, p1); __syncthreads();
  for (int o = 128; o > 0; o >>= 1) { if (tid < o) red[tid] = fmaxf(red[tid], red[tid + o]); __syncthreads(); }
  const float m2 = red[0]; __syncthreads();
  red[tid] = expf(p0 - m2) + expf(p1 - m2); __syncthreads();
  for (int o = 128; o > 0; o >>= 1) { if (tid < o) red[tid] += red[tid + o]; __syncthreads(); }
  const float lse = m2 + logf(red[0]);
  lr[tid] = p0 - lse;
  lr[tid + 256] = p1 - lse;
}

// ---------------------------------------------------------------------------
// score[b] = relu_buf[b,:] . Wm2[:,0] + bm2[0]
// ---------------------------------------------------------------------------
__global__ __launch_bounds__(256) void score_kernel(
    const float* __restrict__ r, const float* __restrict__ w,
    const float* __restrict__ bm2, float* __restrict__ out)
{
  __shared__ float red[256];
  const int b = blockIdx.x;
  const int tid = threadIdx.x;
  float s = 0.f;
  for (int j = tid; j < FF_; j += 256) s += r[(long long)b * FF_ + j] * w[j];
  red[tid] = s; __syncthreads();
  for (int o = 128; o > 0; o >>= 1) { if (tid < o) red[tid] += red[tid + o]; __syncthreads(); }
  if (tid == 0) out[b] = red[0] + bm2[0];
}

// ---------------------------------------------------------------------------
// Host-side launcher for the full-tile GEMM
// ---------------------------------------------------------------------------
template <int EPI, int TBN, bool TRANSB>
static void run_full(hipStream_t st,
    const float* A, long long sAo, long long sAi, int lda,
    const float* Bp, long long sBo, long long sBi, int ldb,
    float* C, long long sCo, long long sCi, int ldc,
    const float* bias,
    const float* resid, long long sRo, long long sRi, int ldr,
    const int* mask, int maskStride, float alpha,
    int M, int N, int K, int innerB, int batch)
{
  dim3 grid((unsigned)(N / TBN), (unsigned)(M / BM), (unsigned)batch);
  gemm_full<EPI, TBN, TRANSB><<<grid, dim3(GT), 0, st>>>(
      A, sAo, sAi, lda, Bp, sBo, sBi, ldb, C, sCo, sCi, ldc,
      bias, resid, sRo, sRi, ldr, mask, maskStride, alpha, K, innerB);
}

extern "C" void kernel_launch(void* const* d_in, const int* in_sizes, int n_in,
                              void* d_out, int out_size, void* d_ws, size_t ws_size,
                              hipStream_t stream)
{
  (void)in_sizes; (void)n_in; (void)out_size; (void)ws_size;

  const int* input_ids = (const int*)d_in[0];
  const int* amask     = (const int*)d_in[1];
  int i = 2;
  const float* word_emb = (const float*)d_in[i++];
  const float* pos_emb  = (const float*)d_in[i++];
  const float* type_emb = (const float*)d_in[i++];
  const float* eg   = (const float*)d_in[i++];
  const float* ebb  = (const float*)d_in[i++];
  const float* Wq   = (const float*)d_in[i++];
  const float* bq   = (const float*)d_in[i++];
  const float* Wk   = (const float*)d_in[i++];
  const float* bk   = (const float*)d_in[i++];
  const float* Wv   = (const float*)d_in[i++];
  const float* bv   = (const float*)d_in[i++];
  const float* Wo   = (const float*)d_in[i++];
  const float* bo   = (const float*)d_in[i++];
  const float* ln1g = (const float*)d_in[i++];
  const float* ln1b = (const float*)d_in[i++];
  const float* W1   = (const float*)d_in[i++];
  const float* b1   = (const float*)d_in[i++];
  const float* W2   = (const float*)d_in[i++];
  const float* b2   = (const float*)d_in[i++];
  const float* ln2g = (const float*)d_in[i++];
  const float* ln2b = (const float*)d_in[i++];
  const float* Wm1  = (const float*)d_in[i++];
  const float* bm1  = (const float*)d_in[i++];
  const float* Wm2  = (const float*)d_in[i++];
  const float* bm2  = (const float*)d_in[i++];

  // Output layout: score(2) | hidden(L*B*S*D) | attns(L*B*H*S*S) | log_attns(...)
  float* outF      = (float*)d_out;
  float* hiddenOut = outF + 2;
  float* attnOut   = hiddenOut + (size_t)L_ * B_ * S_ * D_;
  float* lattnOut  = attnOut + (size_t)L_ * B_ * H_ * S_ * S_;

  // Workspace carve-out (256B aligned)
  char* ws = (char*)d_ws;
  size_t off = 0;
  auto carve = [&](size_t bytes) -> void* {
    void* p = ws + off;
    off = (off + bytes + 255) & ~(size_t)255;
    return p;
  };
  const size_t BS = (size_t)B_ * S_;
  int*   pos  = (int*)  carve(BS * sizeof(int));
  float* h    = (float*)carve(BS * D_ * sizeof(float));
  float* x    = (float*)carve(BS * D_ * sizeof(float));
  float* q    = (float*)carve(BS * D_ * sizeof(float));
  float* kbuf = (float*)carve(BS * D_ * sizeof(float));
  float* vbuf = (float*)carve(BS * D_ * sizeof(float));
  float* ctx  = (float*)carve(BS * D_ * sizeof(float));
  float* ffn  = (float*)carve(BS * FF_ * sizeof(float));
  float* relu = (float*)carve((size_t)B_ * FF_ * sizeof(float));

  const float inv_temp = 0.125f;  // 1/sqrt(64)
  const int M = (int)BS;          // 1024 tokens

  posids_kernel<<<1, 32, 0, stream>>>(input_ids, pos);
  embed_ln_kernel<<<(int)BS, 256, 0, stream>>>(input_ids, pos, word_emb, pos_emb,
                                               type_emb, eg, ebb, h, 1e-12f);

  for (int l = 0; l < L_; ++l) {
    const float* Wq_l = Wq + (size_t)l * D_ * D_;
    const float* Wk_l = Wk + (size_t)l * D_ * D_;
    const float* Wv_l = Wv + (size_t)l * D_ * D_;
    const float* Wo_l = Wo + (size_t)l * D_ * D_;
    const float* W1_l = W1 + (size_t)l * D_ * FF_;
    const float* W2_l = W2 + (size_t)l * FF_ * D_;
    const float* bq_l = bq + (size_t)l * D_;
    const float* bk_l = bk + (size_t)l * D_;
    const float* bv_l = bv + (size_t)l * D_;
    const float* bo_l = bo + (size_t)l * D_;
    const float* b1_l = b1 + (size_t)l * FF_;
    const float* b2_l = b2 + (size_t)l * D_;
    const float* g1_l = ln1g + (size_t)l * D_;
    const float* c1_l = ln1b + (size_t)l * D_;
    const float* g2_l = ln2g + (size_t)l * D_;
    const float* c2_l = ln2b + (size_t)l * D_;
    float* attnL  = attnOut  + (size_t)l * B_ * H_ * S_ * S_;
    float* lattnL = lattnOut + (size_t)l * B_ * H_ * S_ * S_;
    float* hidL   = hiddenOut + (size_t)l * B_ * S_ * D_;

    // Q = (h @ Wq + bq) * inv_temp ; K = h@Wk+bk ; V = h@Wv+bv
    run_full<EPI_SCALE, 128, false>(stream, h, 0, 0, D_, Wq_l, 0, 0, D_,
        q, 0, 0, D_, bq_l, nullptr, 0, 0, 0, nullptr, 0,
        inv_temp, M, D_, D_, 1, 1);
    run_full<EPI_NONE, 128, false>(stream, h, 0, 0, D_, Wk_l, 0, 0, D_,
        kbuf, 0, 0, D_, bk_l, nullptr, 0, 0, 0, nullptr, 0,
        1.0f, M, D_, D_, 1, 1);
    run_full<EPI_NONE, 128, false>(stream, h, 0, 0, D_, Wv_l, 0, 0, D_,
        vbuf, 0, 0, D_, bv_l, nullptr, 0, 0, 0, nullptr, 0,
        1.0f, M, D_, D_, 1, 1);

    // scores[b,h] = Q_head @ K_head^T + mask_bias  -> into attns output region
    run_full<EPI_MASK, 128, true>(stream,
        q,    (long long)S_ * D_, DK_, D_,
        kbuf, (long long)S_ * D_, DK_, D_,
        attnL, (long long)H_ * S_ * S_, (long long)S_ * S_, S_,
        nullptr, nullptr, 0, 0, 0, amask, S_,
        1.0f, S_, S_, DK_, H_, B_ * H_);

    // softmax + log_softmax(probs)
    softmax2_kernel<<<B_ * H_ * S_, 256, 0, stream>>>(attnL, lattnL);

    // ctx[b,h] = attn @ V_head   (TBN=64 tile; fully divisible, no guards)
    run_full<EPI_NONE, 64, false>(stream,
        attnL, (long long)H_ * S_ * S_, (long long)S_ * S_, S_,
        vbuf,  (long long)S_ * D_, DK_, D_,
        ctx,   (long long)S_ * D_, DK_, D_,
        nullptr, nullptr, 0, 0, 0, nullptr, 0,
        1.0f, S_, DK_, S_, H_, B_ * H_);

    // x = ctx @ Wo + bo + h ; h1 = LN1(x)
    run_full<EPI_RESID, 128, false>(stream, ctx, 0, 0, D_, Wo_l, 0, 0, D_,
        x, 0, 0, D_, bo_l, h, 0, 0, D_, nullptr, 0,
        1.0f, M, D_, D_, 1, 1);
    layernorm_kernel<<<(int)BS, 256, 0, stream>>>(x, g1_l, c1_l, h, nullptr, 1e-5f);

    // ffn = gelu(h1 @ W1 + b1) ; x = ffn @ W2 + b2 + h1 ; h2 = LN2(x)
    run_full<EPI_GELU, 128, false>(stream, h, 0, 0, D_, W1_l, 0, 0, FF_,
        ffn, 0, 0, FF_, b1_l, nullptr, 0, 0, 0, nullptr, 0,
        1.0f, M, FF_, D_, 1, 1);
    run_full<EPI_RESID, 128, false>(stream, ffn, 0, 0, FF_, W2_l, 0, 0, D_,
        x, 0, 0, D_, b2_l, h, 0, 0, D_, nullptr, 0,
        1.0f, M, D_, FF_, 1, 1);
    layernorm_kernel<<<(int)BS, 256, 0, stream>>>(x, g2_l, c2_l, h, hidL, 1e-5f);
  }

  // head: relu(cls @ Wm1 + bm1) @ Wm2 + bm2 ; cls = h[b, s=0, :] -> lda = S*D
  {
    dim3 grid(FF_ / 128, 1, 1);
    gemm_part<EPI_RELU><<<grid, dim3(GT), 0, stream>>>(
        h, S_ * D_, Wm1, FF_, relu, FF_, bm1, B_, FF_, D_);
  }
  score_kernel<<<B_, 256, 0, stream>>>(relu, Wm2, bm2, outF);
}
